// GAT_76012331205027
// MI455X (gfx1250) — compile-verified, matrix operand order checked by reading, not run on hardware
//
#include <hip/hip_runtime.h>

typedef __bf16 bf16;
typedef __attribute__((ext_vector_type(16))) __bf16 v16bf;
typedef __attribute__((ext_vector_type(8)))  __bf16 v8bf;
typedef __attribute__((ext_vector_type(8)))  float  v8f;

#define NNODES 50000

// ---------------------------------------------------------------- utilities

__global__ void fill_f32(float* __restrict__ p, long long n, float val) {
  long long i = (long long)blockIdx.x * blockDim.x + threadIdx.x;
  if (i < n) p[i] = val;
}

__global__ void degree_kernel(const long long* __restrict__ src,
                              const long long* __restrict__ dst,
                              float* __restrict__ deg_out,
                              float* __restrict__ deg_in, long long E) {
  long long e = (long long)blockIdx.x * blockDim.x + threadIdx.x;
  if (e < E) {
    atomicAdd(&deg_out[src[e]], 1.0f);
    atomicAdd(&deg_in[dst[e]], 1.0f);
  }
}

__global__ void rsqrt_clamp_kernel(float* __restrict__ p, int n) {
  int i = blockIdx.x * blockDim.x + threadIdx.x;
  if (i < n) {
    float d = p[i];
    p[i] = rsqrtf(d < 1.0f ? 1.0f : d);
  }
}

// W is KxN row-major fp32; emit W^T as NxK row-major bf16 so the GEMM's B
// fragments load with the same contiguous A-style layout as the A fragments.
__global__ void cast_transpose_kernel(const float* __restrict__ in,
                                      bf16* __restrict__ out, int K, int N) {
  int i = blockIdx.x * blockDim.x + threadIdx.x;
  if (i < K * N) {
    int k = i / N, n = i % N;
    out[(long long)n * K + k] = (bf16)in[i];
  }
}

// out[i] = bf16(x[i] * rs[row])   (layer-1 input only)
__global__ void scale_cast_kernel(const float* __restrict__ x,
                                  const float* __restrict__ rs,
                                  bf16* __restrict__ out, long long n, int dshift) {
  long long i = (long long)blockIdx.x * blockDim.x + threadIdx.x;
  if (i < n) {
    out[i] = (bf16)(x[i] * rs[i >> dshift]);
  }
}

// ---------------------------------------------------------------- WMMA GEMM

// ISA 7.12.2 16-bit A-matrix 16x32 layout: lane<16 -> row=lane, K = 0..7,16..23;
// lane>=16 -> row=lane-16, K = 8..15,24..31. Each half is 8 contiguous bf16
// (16B) -> two b128 loads per fragment.
__device__ __forceinline__ v16bf load_frag(const bf16* __restrict__ p, int ld, int lane) {
  int half = (lane >> 4) & 1;
  int r = lane & 15;
  const bf16* base = p + (long long)r * ld + half * 8;
  v8bf lo = *(const v8bf*)(base);
  v8bf hi = *(const v8bf*)(base + 16);
  return __builtin_shufflevector(lo, hi, 0,1,2,3,4,5,6,7,8,9,10,11,12,13,14,15);
}

// C[M,N] = A[M,K] @ Bt[N,K]^T, f32 accumulate. One 16x16 tile per wave,
// 8 waves per block strided along N. M % 16 == 0, K % 32 == 0, N % 16 == 0.
__global__ void __launch_bounds__(256)
wmma_gemm_kernel(const bf16* __restrict__ A, const bf16* __restrict__ Bt,
                 float* __restrict__ C, int M, int N, int K) {
  int lane  = threadIdx.x & 31;
  int wave  = threadIdx.x >> 5;
  int ntile = blockIdx.y * 8 + wave;
  if (ntile * 16 >= N) return;                   // wave-uniform: EXEC stays full
  long long row0 = (long long)blockIdx.x * 16;
  const bf16* Ap = A + row0 * K;
  const bf16* Bp = Bt + (long long)ntile * 16 * K;

  v8f acc = {};
  for (int k = 0; k < K; k += 32) {
    v16bf a = load_frag(Ap + k, K, lane);
    v16bf b = load_frag(Bp + k, K, lane);
    acc = __builtin_amdgcn_wmma_f32_16x16x32_bf16(
        /*neg_a=*/false, a, /*neg_b=*/false, b,
        /*c_mod=*/(short)0, acc, /*reuse_a=*/false, /*reuse_b=*/false);
  }
  // C layout: VGPR v -> row = v + 8*(lane>=16), col = lane&15
  int col = ntile * 16 + (lane & 15);
  long long rbase = row0 + ((lane >> 4) << 3);
  #pragma unroll
  for (int v = 0; v < 8; ++v)
    C[(rbase + v) * N + col] = acc[v];
}

// ---------------------------------------------------------------- scatter

// One thread per (edge, 4-dim chunk): b128 gather + 4x global_atomic_add_f32.
__global__ void scatter_add_vec4_kernel(const float* __restrict__ H,
                                        const long long* __restrict__ src,
                                        const long long* __restrict__ dst,
                                        float* __restrict__ agg,
                                        long long E, int dshift) {
  int cshift = dshift - 2;                       // chunks of 4 floats
  long long idx = (long long)blockIdx.x * blockDim.x + threadIdx.x;
  long long total = E << cshift;
  if (idx >= total) return;
  long long e = idx >> cshift;
  int c = (int)(idx & ((1LL << cshift) - 1));
  long long s = src[e], t = dst[e];
  const float4 hv = *(const float4*)(H + (s << dshift) + (c << 2));
  float* a = agg + (t << dshift) + (c << 2);
  atomicAdd(a + 0, hv.x);
  atomicAdd(a + 1, hv.y);
  atomicAdd(a + 2, hv.z);
  atomicAdd(a + 3, hv.w);
}

// ---------------------------------------------------------------- fused post

// v = in[i]; v *= rs_in[row] (if given); v += bias[d]; relu (if asked).
// Optionally writes f32 copy (h_last / final out) and/or the next GEMM's
// bf16 input, pre-scaled by rs_out[row] (if given). Fuses epilogue+cast so
// intermediate activations never round-trip HBM in f32.
__global__ void fused_post_kernel(const float* __restrict__ in,
                                  const float* __restrict__ rs_in,
                                  const float* __restrict__ bias,
                                  const float* __restrict__ rs_out,
                                  bf16* __restrict__ out_bf,
                                  float* __restrict__ out_f32,
                                  long long n, int dshift, int do_relu) {
  long long i = (long long)blockIdx.x * blockDim.x + threadIdx.x;
  if (i >= n) return;
  long long row = i >> dshift;
  int d = (int)(i & ((1LL << dshift) - 1));
  float v = in[i];
  if (rs_in) v *= rs_in[row];
  v += bias[d];
  if (do_relu) v = v > 0.0f ? v : 0.0f;
  if (out_f32) out_f32[i] = v;
  if (out_bf) {
    float s = rs_out ? rs_out[row] : 1.0f;
    out_bf[i] = (bf16)(v * s);
  }
}

// ---------------------------------------------------------------- launch

extern "C" void kernel_launch(void* const* d_in, const int* in_sizes, int n_in,
                              void* d_out, int out_size, void* d_ws, size_t ws_size,
                              hipStream_t stream) {
  const float*     x   = (const float*)d_in[0];
  const long long* ei  = (const long long*)d_in[1];   // int64 [2, E]
  const float* W1 = (const float*)d_in[2];  const float* b1 = (const float*)d_in[3];
  const float* W2 = (const float*)d_in[4];  const float* b2 = (const float*)d_in[5];
  const float* W3 = (const float*)d_in[6];  const float* b3 = (const float*)d_in[7];
  const float* mW1= (const float*)d_in[8];  const float* mb1= (const float*)d_in[9];
  const float* mW2= (const float*)d_in[10]; const float* mb2= (const float*)d_in[11];

  const long long E = in_sizes[1] / 2;
  const long long* src = ei;
  const long long* dst = ei + E;
  const long long NN = NNODES;

  float* out_final = (float*)d_out;                   // 50000 x 64
  float* h_last    = out_final + NN * 64;             // 50000 x 128

  char* w = (char*)d_ws;
  auto carve = [&](size_t bytes) {
    void* p = (void*)w; w += (bytes + 255) & ~(size_t)255; return p;
  };
  float* rs_out = (float*)carve(NN * 4);
  float* rs_in  = (float*)carve(NN * 4);
  bf16* W1t = (bf16*)carve(256 * 256 * 2);
  bf16* W2t = (bf16*)carve(256 * 256 * 2);
  bf16* W3t = (bf16*)carve(256 * 128 * 2);
  bf16* M1t = (bf16*)carve(128 * 64 * 2);
  bf16* M2t = (bf16*)carve(64 * 64 * 2);
  bf16*  XS  = (bf16*) carve((size_t)NN * 256 * 2);
  float* H   = (float*)carve((size_t)NN * 256 * 4);
  float* AGG = (float*)carve((size_t)NN * 256 * 4);

  const int T = 256;
  auto nb = [](long long n, int t) { return (unsigned)((n + t - 1) / t); };

  // --- degrees -> rsqrt(clip(deg,1))
  fill_f32<<<nb(NN, T), T, 0, stream>>>(rs_out, NN, 0.0f);
  fill_f32<<<nb(NN, T), T, 0, stream>>>(rs_in,  NN, 0.0f);
  degree_kernel<<<nb(E, T), T, 0, stream>>>(src, dst, rs_out, rs_in, E);
  rsqrt_clamp_kernel<<<nb(NN, T), T, 0, stream>>>(rs_out, (int)NN);
  rsqrt_clamp_kernel<<<nb(NN, T), T, 0, stream>>>(rs_in,  (int)NN);

  // --- weights -> transposed bf16
  cast_transpose_kernel<<<nb(256*256, T), T, 0, stream>>>(W1,  W1t, 256, 256);
  cast_transpose_kernel<<<nb(256*256, T), T, 0, stream>>>(W2,  W2t, 256, 256);
  cast_transpose_kernel<<<nb(256*128, T), T, 0, stream>>>(W3,  W3t, 256, 128);
  cast_transpose_kernel<<<nb(128*64,  T), T, 0, stream>>>(mW1, M1t, 128, 64);
  cast_transpose_kernel<<<nb(64*64,   T), T, 0, stream>>>(mW2, M2t, 64, 64);

  dim3 g256(NNODES / 16, 2);   // N=256 -> 16 n-tiles, 8 waves/block
  dim3 g128(NNODES / 16, 1);   // N=128
  dim3 g64 (NNODES / 16, 1);   // N=64 (4 of 8 waves active)

  // --- GCN layer 1 (in 256 -> out 256): XS <- bf16(x * rs_out)
  scale_cast_kernel<<<nb(NN*256, T), T, 0, stream>>>(x, rs_out, XS, NN*256, 8);
  wmma_gemm_kernel<<<g256, T, 0, stream>>>(XS, W1t, H, (int)NN, 256, 256);
  fill_f32<<<nb(NN*256, T), T, 0, stream>>>(AGG, NN*256, 0.0f);
  scatter_add_vec4_kernel<<<nb(E << 6, T), T, 0, stream>>>(H, src, dst, AGG, E, 8);
  // XS <- bf16(relu(AGG*rs_in + b1) * rs_out)   (layer-2 GEMM input)
  fused_post_kernel<<<nb(NN*256, T), T, 0, stream>>>(AGG, rs_in, b1, rs_out,
                                                     XS, nullptr, NN*256, 8, 1);

  // --- GCN layer 2 (256 -> 256)
  wmma_gemm_kernel<<<g256, T, 0, stream>>>(XS, W2t, H, (int)NN, 256, 256);
  fill_f32<<<nb(NN*256, T), T, 0, stream>>>(AGG, NN*256, 0.0f);
  scatter_add_vec4_kernel<<<nb(E << 6, T), T, 0, stream>>>(H, src, dst, AGG, E, 8);
  fused_post_kernel<<<nb(NN*256, T), T, 0, stream>>>(AGG, rs_in, b2, rs_out,
                                                     XS, nullptr, NN*256, 8, 1);

  // --- GCN layer 3 (256 -> 128): writes f32 h_last AND bf16 MLP input
  wmma_gemm_kernel<<<g128, T, 0, stream>>>(XS, W3t, H, (int)NN, 128, 256);
  fill_f32<<<nb(NN*128, T), T, 0, stream>>>(AGG, NN*128, 0.0f);
  scatter_add_vec4_kernel<<<nb(E << 5, T), T, 0, stream>>>(H, src, dst, AGG, E, 7);
  fused_post_kernel<<<nb(NN*128, T), T, 0, stream>>>(AGG, rs_in, b3, nullptr,
                                                     XS, h_last, NN*128, 7, 1);

  // --- MLP: relu(h_last @ mW1 + mb1) @ mW2 + mb2
  wmma_gemm_kernel<<<g64, T, 0, stream>>>(XS, M1t, H, (int)NN, 64, 128);
  fused_post_kernel<<<nb(NN*64, T), T, 0, stream>>>(H, nullptr, mb1, nullptr,
                                                    XS, nullptr, NN*64, 6, 1);
  wmma_gemm_kernel<<<g64, T, 0, stream>>>(XS, M2t, H, (int)NN, 64, 64);
  fused_post_kernel<<<nb(NN*64, T), T, 0, stream>>>(H, nullptr, mb2, nullptr,
                                                    nullptr, out_final, NN*64, 6, 0);
}
